// LSTM_NLI_53154515256227
// MI455X (gfx1250) — matrix operand order repaired and checked by reading, not compiled, not run on hardware
//
#include <hip/hip_runtime.h>
#include <hip/hip_bf16.h>

// ---------------------------------------------------------------------------
// LSTM sentence embedding for MI455X (gfx1250, wave32, WMMA).
// B=64, L=128, V=100000, E=300 (pad->320), H=512, G=4H=2048, O=3.
//
// FLOPs: 10.7 GF (input proj) + 17.2 GF (recurrence) in bf16 WMMA w/ f32 acc.
// Bytes: ~90 MB total -> compute/latency bound; weights stay L2-resident.
// ---------------------------------------------------------------------------

typedef __attribute__((ext_vector_type(16))) __bf16 v16bf;
typedef __attribute__((ext_vector_type(8)))  __bf16 v8bf;
typedef __attribute__((ext_vector_type(8)))  float  v8f;

#define Bn 64
#define Ln 128
#define En 300
#define EP 320     // padded K for phase-1 GEMM (10 k-tiles of 32)
#define Hn 512
#define Gn 2048    // 4*H
#define On 3
#define NROW (Bn * Ln)   // 8192 token rows

// ---------------- fragment loaders ----------------
// A-fragment (16x32 bf16, MxK), source row-major with given stride.
// ISA layout: lanes 0-15 -> row M=lane, K = {0..7, 16..23};
//             lanes 16-31 -> row M=lane-16, K = {8..15, 24..31}.
__device__ inline v16bf load_a_frag(const __bf16* base, int stride,
                                    int row0, int kbase, int lane) {
    int row = row0 + (lane & 15);
    int k0  = kbase + ((lane & 16) ? 8 : 0);
    const __bf16* p = base + (size_t)row * stride + k0;
    v8bf lo = *(const v8bf*)(p);        // K offsets +0..7  (or +8..15)
    v8bf hi = *(const v8bf*)(p + 16);   // K offsets +16..23 (or +24..31)
    v16bf a;
#pragma unroll
    for (int i = 0; i < 8; ++i) { a[i] = lo[i]; a[i + 8] = hi[i]; }
    return a;
}

// B-fragment (32x16 bf16, KxN) from pre-packed tile: lane-contiguous 32B.
__device__ inline v16bf load_b_frag(const __bf16* tile, int lane) {
    const v8bf* p = (const v8bf*)(tile + lane * 16);
    v8bf lo = p[0];
    v8bf hi = p[1];
    v16bf b;
#pragma unroll
    for (int i = 0; i < 8; ++i) { b[i] = lo[i]; b[i + 8] = hi[i]; }
    return b;
}

// ---------------- weight packing ----------------
// Pack f32 weight W[G][K] into bf16 B-fragments. Tile (nt, kt) holds
// B[k][n] = W[nt*16+n][kt*32+k]; element index = lane*16 + s with
// n = lane%16, k = 8*(lane/16) + (s&7) + 16*(s/8). Zero-pad k >= K.
__global__ void pack_w_kernel(const float* __restrict__ W, __bf16* __restrict__ dst,
                              int K, int KT, int total) {
    int idx = blockIdx.x * blockDim.x + threadIdx.x;
    if (idx >= total) return;
    int tile = idx >> 9;          // /512
    int e    = idx & 511;
    int lane = e >> 4;            // fragment lane: n = lane%16, k-half = lane/16
    int s    = e & 15;
    int n    = lane & 15;
    int nt   = tile / KT;
    int kt   = tile % KT;
    int k    = kt * 32 + (s & 7) + ((lane >> 4) << 3) + ((s & 8) << 1);
    int g    = nt * 16 + n;
    float v  = (k < K) ? W[(size_t)g * K + k] : 0.0f;
    dst[idx] = (__bf16)v;
}

// ---------------- embedding gather + bf16 convert ----------------
__global__ void gather_kernel(const int* __restrict__ sentences,
                              const float* __restrict__ embed,
                              __bf16* __restrict__ Xbf) {
    int idx = blockIdx.x * blockDim.x + threadIdx.x;   // over NROW*EP
    if (idx >= NROW * EP) return;
    int row = idx / EP;
    int e   = idx - row * EP;
    float v = 0.0f;
    if (e < En) {
        int tok = sentences[row];
        v = embed[(size_t)tok * En + e];
    }
    Xbf[idx] = (__bf16)v;
}

// ---------------- phase 1: xg = X * W_ih^T + (b_ih + b_hh) ----------------
// One 16x16 output tile per wave. M-tiles = 512, N-tiles = 128, K-tiles = 10.
__global__ void gemm_xw_kernel(const __bf16* __restrict__ Xbf,
                               const __bf16* __restrict__ Wih_p,
                               const float* __restrict__ b_ih,
                               const float* __restrict__ b_hh,
                               float* __restrict__ xg) {
    int wave = blockIdx.x * (blockDim.x >> 5) + (threadIdx.x >> 5);
    int lane = threadIdx.x & 31;
    const int NT = Gn / 16;           // 128
    int mt = wave / NT;               // 0..511
    int nt = wave - mt * NT;          // 0..127
    if (mt >= NROW / 16) return;

    v8f acc = {};
    for (int kt = 0; kt < EP / 32; ++kt) {
        v16bf a = load_a_frag(Xbf, EP, mt * 16, kt * 32, lane);
        v16bf b = load_b_frag(Wih_p + ((size_t)(nt * (EP / 32) + kt) << 9), lane);
        acc = __builtin_amdgcn_wmma_f32_16x16x32_bf16(
            false, a, false, b, (short)0, acc, false, false);
    }
    int col  = nt * 16 + (lane & 15);
    float bias = b_ih[col] + b_hh[col];
    int mhi  = (lane & 16) ? 8 : 0;
#pragma unroll
    for (int r = 0; r < 8; ++r) {
        int row = mt * 16 + r + mhi;                  // token row = b*L + l
        xg[(size_t)row * Gn + col] = acc[r] + bias;
    }
}

// ---------------- phase 2 GEMM: gates = xg[t] + h * W_hh^T ----------------
// M-tiles = 4 (B=64), N-tiles = 128, K-tiles = 16 (H=512). 512 tiles/step.
__global__ void gemm_hw_kernel(const __bf16* __restrict__ hbf,
                               const __bf16* __restrict__ Whh_p,
                               const float* __restrict__ xg,
                               float* __restrict__ gates,
                               int t) {
    int wave = blockIdx.x * (blockDim.x >> 5) + (threadIdx.x >> 5);
    int lane = threadIdx.x & 31;
    const int NT = Gn / 16;           // 128
    int mt = wave / NT;               // 0..3
    int nt = wave - mt * NT;
    if (mt >= Bn / 16) return;

    v8f acc = {};
    for (int kt = 0; kt < Hn / 32; ++kt) {
        v16bf a = load_a_frag(hbf, Hn, mt * 16, kt * 32, lane);
        v16bf b = load_b_frag(Whh_p + ((size_t)(nt * (Hn / 32) + kt) << 9), lane);
        acc = __builtin_amdgcn_wmma_f32_16x16x32_bf16(
            false, a, false, b, (short)0, acc, false, false);
    }
    int col = nt * 16 + (lane & 15);
    int mhi = (lane & 16) ? 8 : 0;
#pragma unroll
    for (int r = 0; r < 8; ++r) {
        int b   = mt * 16 + r + mhi;                  // batch index
        size_t xrow = (size_t)(b * Ln + t) * Gn;      // xg is [b*L + l][G]
        gates[(size_t)b * Gn + col] = acc[r] + xg[xrow + col];
    }
}

// ---------------- LSTM cell elementwise ----------------
__device__ inline float sigf(float x) { return 1.0f / (1.0f + __expf(-x)); }
__device__ inline float tanhfast(float x) {
    float e = __expf(2.0f * x);
    return (e - 1.0f) / (e + 1.0f);
}

__global__ void cell_kernel(const float* __restrict__ gates,
                            float* __restrict__ c,
                            float* __restrict__ h,
                            __bf16* __restrict__ hbf) {
    int idx = blockIdx.x * blockDim.x + threadIdx.x;   // over B*H
    if (idx >= Bn * Hn) return;
    int b = idx / Hn;
    int j = idx - b * Hn;
    const float* gb = gates + (size_t)b * Gn;
    float i = sigf(gb[j]);
    float f = sigf(gb[Hn + j]);
    float g = tanhfast(gb[2 * Hn + j]);
    float o = sigf(gb[3 * Hn + j]);
    float cn = f * c[idx] + i * g;
    float hn = o * tanhfast(cn);
    c[idx]   = cn;
    h[idx]   = hn;
    hbf[idx] = (__bf16)hn;
}

__global__ void init_state_kernel(float* __restrict__ c, __bf16* __restrict__ hbf) {
    int idx = blockIdx.x * blockDim.x + threadIdx.x;
    if (idx >= Bn * Hn) return;
    c[idx] = 0.0f;
    hbf[idx] = (__bf16)0.0f;
}

// ---------------- final projection: out = h * W_out^T + b_out ----------------
__global__ void proj_kernel(const float* __restrict__ h,
                            const float* __restrict__ W_out,
                            const float* __restrict__ b_out,
                            float* __restrict__ out) {
    int idx = blockIdx.x * blockDim.x + threadIdx.x;   // over B*O
    if (idx >= Bn * On) return;
    int b = idx / On;
    int o = idx - b * On;
    float s = b_out[o];
    const float* hp = h + (size_t)b * Hn;
    const float* wp = W_out + (size_t)o * Hn;
    for (int k = 0; k < Hn; ++k) s += hp[k] * wp[k];
    out[idx] = s;
}

// ---------------------------------------------------------------------------
extern "C" void kernel_launch(void* const* d_in, const int* in_sizes, int n_in,
                              void* d_out, int out_size, void* d_ws, size_t ws_size,
                              hipStream_t stream) {
    const int*   sentences = (const int*)  d_in[0];
    const float* embed     = (const float*)d_in[1];
    const float* W_ih      = (const float*)d_in[2];
    const float* W_hh      = (const float*)d_in[3];
    const float* b_ih      = (const float*)d_in[4];
    const float* b_hh      = (const float*)d_in[5];
    const float* W_out     = (const float*)d_in[6];
    const float* b_out     = (const float*)d_in[7];
    float* out = (float*)d_out;

    // Workspace carve-up (~77 MB total).
    char* ws = (char*)d_ws;
    __bf16* Xbf   = (__bf16*)ws;  ws += (size_t)NROW * EP * 2;              // 5.24 MB
    __bf16* Wih_p = (__bf16*)ws;  ws += (size_t)(Gn/16) * (EP/32) * 512 * 2; // 1.31 MB
    __bf16* Whh_p = (__bf16*)ws;  ws += (size_t)(Gn/16) * (Hn/32) * 512 * 2; // 2.10 MB
    float*  xg    = (float*) ws;  ws += (size_t)NROW * Gn * 4;              // 67.1 MB
    float*  gates = (float*) ws;  ws += (size_t)Bn * Gn * 4;                // 0.52 MB
    float*  hf    = (float*) ws;  ws += (size_t)Bn * Hn * 4;
    float*  cf    = (float*) ws;  ws += (size_t)Bn * Hn * 4;
    __bf16* hbf   = (__bf16*)ws;  ws += (size_t)Bn * Hn * 2;

    // --- setup: weight packing, gather, state init ---
    {
        int total = (Gn/16) * (EP/32) * 512;
        pack_w_kernel<<<(total + 255) / 256, 256, 0, stream>>>(W_ih, Wih_p, En, EP/32, total);
    }
    {
        int total = (Gn/16) * (Hn/32) * 512;
        pack_w_kernel<<<(total + 255) / 256, 256, 0, stream>>>(W_hh, Whh_p, Hn, Hn/32, total);
    }
    gather_kernel<<<(NROW * EP + 255) / 256, 256, 0, stream>>>(sentences, embed, Xbf);
    init_state_kernel<<<(Bn * Hn + 255) / 256, 256, 0, stream>>>(cf, hbf);

    // --- phase 1: all-timestep input projection via bf16 WMMA ---
    {
        int tiles  = (NROW / 16) * (Gn / 16);   // 65536 waves
        int blocks = tiles / 8;                 // 8 waves per 256-thread block
        gemm_xw_kernel<<<blocks, 256, 0, stream>>>(Xbf, Wih_p, b_ih, b_hh, xg);
    }

    // --- phase 2: sequential recurrence, WMMA GEMM + cell per step ---
    for (int t = 0; t < Ln; ++t) {
        int tiles  = (Bn / 16) * (Gn / 16);     // 512 waves
        int blocks = tiles / 8;                 // 64 blocks
        gemm_hw_kernel<<<blocks, 256, 0, stream>>>(hbf, Whh_p, xg, gates, t);
        cell_kernel<<<(Bn * Hn + 255) / 256, 256, 0, stream>>>(gates, cf, hf, hbf);
    }

    // --- final projection ---
    proj_kernel<<<1, 256, 0, stream>>>(hf, W_out, b_out, out);
}
